// BandSplit_79139067396476
// MI455X (gfx1250) — matrix operand order repaired and checked by reading, not compiled, not run on hardware
//
#include <hip/hip_runtime.h>

#define BANDS    128
#define D_OUTF   128
#define T_LEN    512
#define F_BINS   1025
#define TILE_M   64
#define WC_PAD   160          // padded W*C (multiple of 32); actual i1 ~ 134
#define WMAX     (WC_PAD / 2)
#define NTHREADS 256

typedef __attribute__((ext_vector_type(16))) __bf16 v16bf;
typedef __attribute__((ext_vector_type(8)))  __bf16 v8bf;
typedef __attribute__((ext_vector_type(8)))  float  v8f;

__global__ void bs_zero_kernel(float* __restrict__ out, int n) {
  int i = blockIdx.x * blockDim.x + threadIdx.x;
  int stride = gridDim.x * blockDim.x;
  for (; i < n; i += stride) out[i] = 0.0f;
}

// One-time per launch: convert + transpose weights into bf16 LDS-ready layout in d_ws.
//   wsBpre [k][n][kk]  (kk padded to WC_PAD)   <- pre_w  (k, i1, 128)
//   wsBpost[k][o][d]   (o  padded to WC_PAD)   <- post_w (k, 128, i1)
__global__ void bs_prep_weights(const float* __restrict__ pre_w,
                                const float* __restrict__ post_w,
                                __bf16* __restrict__ wsBpre,
                                __bf16* __restrict__ wsBpost, int i1) {
  const int total = BANDS * D_OUTF * WC_PAD;
  int stride = gridDim.x * blockDim.x;
  for (int e = blockIdx.x * blockDim.x + threadIdx.x; e < total; e += stride) {
    int kk = e % WC_PAD;                 // innermost -> coalesced writes
    int n  = (e / WC_PAD) % D_OUTF;
    int k  = e / (WC_PAD * D_OUTF);
    float v = (kk < i1) ? pre_w[((long)k * i1 + kk) * D_OUTF + n] : 0.0f;
    wsBpre[e] = (__bf16)v;
  }
  for (int e = blockIdx.x * blockDim.x + threadIdx.x; e < total; e += stride) {
    int d = e % D_OUTF;                  // innermost -> coalesced writes
    int o = (e / D_OUTF) % WC_PAD;
    int k = e / (WC_PAD * D_OUTF);
    float v = (o < i1) ? post_w[((long)k * D_OUTF + d) * i1 + o] : 0.0f;
    wsBpost[e] = (__bf16)v;
  }
}

__device__ inline void wait_async0() {
#if __has_builtin(__builtin_amdgcn_s_wait_asynccnt)
  __builtin_amdgcn_s_wait_asynccnt(0);
#else
  asm volatile("s_wait_asynccnt 0x0" ::: "memory");
#endif
}

// ASYNCcnt-tracked global->LDS DMA, 16B per issue (GVS addressing: saddr + vaddr).
__device__ inline void async_copy_b128(unsigned lds_byte_addr, const void* saddr,
                                       unsigned byte_off) {
  asm volatile("global_load_async_to_lds_b128 %0, %1, %2"
               :: "v"(lds_byte_addr), "v"(byte_off), "s"(saddr)
               : "memory");
}

// Build 16x32 bf16 A fragment (contiguous k, matching B):
// lanes 0-15: V0-7 = K[k0..k0+15]
// lanes16-31: V0-7 = K[k0+16..k0+31]
__device__ inline v16bf load_a_frag(const __bf16* row, int k0, int lhi) {
  v8bf lo = *(const v8bf*)(row + k0 + lhi * 16);
  v8bf hi = *(const v8bf*)(row + k0 + lhi * 16 + 8);
  return __builtin_shufflevector(lo, hi, 0,1,2,3,4,5,6,7,8,9,10,11,12,13,14,15);
}

template <bool USE_WS>
__global__ __launch_bounds__(NTHREADS, 1)
void bandsplit_fused(const float* __restrict__ x,
                     const float* __restrict__ pre_w,  const float* __restrict__ pre_b,
                     const float* __restrict__ post_w, const float* __restrict__ post_b,
                     const float* __restrict__ mel,    const float* __restrict__ mask,
                     const float* __restrict__ ola,    const int* __restrict__ idx,
                     const __bf16* __restrict__ wsBpre, const __bf16* __restrict__ wsBpost,
                     float* __restrict__ out, int W, int i1) {
  __shared__ __align__(32) __bf16 sG[TILE_M * WC_PAD];     // gathered tile (m x i)
  __shared__ __align__(32) __bf16 sBpre[D_OUTF * WC_PAD];  // pre_w transposed [n][k]
  __shared__ __align__(32) __bf16 sBpost[WC_PAD * D_OUTF]; // post_w transposed [o][d]
  __shared__ __align__(32) __bf16 sZ[TILE_M * D_OUTF];     // hidden tile (m x d)
  __shared__ float sPreB[D_OUTF];
  __shared__ float sPostB[WC_PAD];
  __shared__ float sMel[WMAX];
  __shared__ float sMask[WMAX];
  __shared__ float sOlaInv[WMAX];
  __shared__ int   sIdx[WMAX];

  const int tid  = threadIdx.x;
  const int k    = blockIdx.y;          // band
  const int row0 = blockIdx.x * TILE_M; // first (b,t) row of the tile

  // ---- stage weights: async DMA of pre-converted bf16 tiles (fast path) ----
  if (USE_WS) {
    const __bf16* bpre  = wsBpre  + (long)k * (D_OUTF * WC_PAD);
    const __bf16* bpost = wsBpost + (long)k * (D_OUTF * WC_PAD);
    const unsigned ldsPre  = (unsigned)(size_t)&sBpre[0];
    const unsigned ldsPost = (unsigned)(size_t)&sBpost[0];
    // 40960 bytes each = 2560 x b128; 256 threads -> 10 issues per array
    #pragma unroll
    for (int j = 0; j < (D_OUTF * WC_PAD * 2) / (16 * NTHREADS); ++j) {
      unsigned off = (unsigned)(tid + j * NTHREADS) * 16u;
      async_copy_b128(ldsPre + off, bpre, off);
      async_copy_b128(ldsPost + off, bpost, off);
    }
  }

  // ---- stage 0: per-band tables + biases ----
  for (int w = tid; w < WMAX; w += NTHREADS) {
    int f = F_BINS - 1; float mm = 0.0f, mk = 0.0f;
    if (w < W) { f = idx[k * W + w]; mk = mask[k * W + w]; mm = mel[k * W + w] * mk; }
    sIdx[w] = f; sMel[w] = mm; sMask[w] = mk;
    sOlaInv[w] = 1.0f / ola[f];
  }
  for (int n = tid; n < D_OUTF; n += NTHREADS) sPreB[n] = pre_b[k * D_OUTF + n];
  for (int o = tid; o < WC_PAD; o += NTHREADS) sPostB[o] = (o < i1) ? post_b[k * i1 + o] : 0.0f;
  __syncthreads();

  // ---- stage 1: gather input tile (+ weights on fallback path) ----
  for (int e = tid; e < TILE_M * WC_PAD; e += NTHREADS) {
    int m = e / WC_PAD, i = e - m * WC_PAD;
    float v = 0.0f;
    if (i < i1) {
      int w = i >> 1, c = i & 1;
      int rg = row0 + m, bb = rg >> 9, tt = rg & (T_LEN - 1);
      v = x[((bb * 2 + c) * T_LEN + tt) * F_BINS + sIdx[w]] * sMel[w];
    }
    sG[e] = (__bf16)v;
  }
  if (!USE_WS) {
    const float* pwk = pre_w + (long)k * i1 * D_OUTF;      // (i1, 128) row-major
    for (int e = tid; e < D_OUTF * WC_PAD; e += NTHREADS) {
      int kk = e >> 7, n = e & (D_OUTF - 1);               // coalesced over n
      float v = (kk < i1) ? pwk[kk * D_OUTF + n] : 0.0f;
      sBpre[n * WC_PAD + kk] = (__bf16)v;                  // transpose -> [n][k]
    }
    const float* qwk = post_w + (long)k * D_OUTF * i1;     // (128, i1) row-major
    for (int e = tid; e < D_OUTF * WC_PAD; e += NTHREADS) {
      int d = e / WC_PAD, o = e - d * WC_PAD;              // coalesced over o
      float v = (o < i1) ? qwk[d * i1 + o] : 0.0f;
      sBpost[o * D_OUTF + d] = (__bf16)v;                  // transpose -> [o][d]
    }
  } else {
    wait_async0();  // all of this wave's DMA landed in LDS
  }
  __syncthreads();

  const int wv   = tid >> 5;
  const int lane = tid & 31;
  const int lrow = lane & 15;
  const int lhi  = lane >> 4;

  // ---- GEMM1: Z(64x128) = G(64xWC_PAD) x pre_w, bias, -> LDS bf16 ----
  {
    const int m0  = (wv & 3) * 16;   // wave's M tile
    const int nb0 = (wv >> 2) * 64;  // wave's N strip (4 tiles of 16)
    v8f acc0 = {}, acc1 = {}, acc2 = {}, acc3 = {};
    for (int k0 = 0; k0 < WC_PAD; k0 += 32) {
      v16bf a  = load_a_frag(sG + (m0 + lrow) * WC_PAD, k0, lhi);
      int koff = k0 + lhi * 16;
      v16bf b0 = *(const v16bf*)(sBpre + (nb0 +  0 + lrow) * WC_PAD + koff);
      v16bf b1 = *(const v16bf*)(sBpre + (nb0 + 16 + lrow) * WC_PAD + koff);
      v16bf b2 = *(const v16bf*)(sBpre + (nb0 + 32 + lrow) * WC_PAD + koff);
      v16bf b3 = *(const v16bf*)(sBpre + (nb0 + 48 + lrow) * WC_PAD + koff);
      acc0 = __builtin_amdgcn_wmma_f32_16x16x32_bf16(false, a, false, b0, (short)0, acc0, false, false);
      acc1 = __builtin_amdgcn_wmma_f32_16x16x32_bf16(false, a, false, b1, (short)0, acc1, false, false);
      acc2 = __builtin_amdgcn_wmma_f32_16x16x32_bf16(false, a, false, b2, (short)0, acc2, false, false);
      acc3 = __builtin_amdgcn_wmma_f32_16x16x32_bf16(false, a, false, b3, (short)0, acc3, false, false);
    }
#pragma unroll
    for (int r = 0; r < 8; ++r) {
      int mm = m0 + r + 8 * lhi;       // C/D layout: VGPR r, lanes 0-15 -> M=r, 16-31 -> M=r+8
      int n0 = nb0 + lrow;
      sZ[mm * D_OUTF + n0     ] = (__bf16)(acc0[r] + sPreB[n0     ]);
      sZ[mm * D_OUTF + n0 + 16] = (__bf16)(acc1[r] + sPreB[n0 + 16]);
      sZ[mm * D_OUTF + n0 + 32] = (__bf16)(acc2[r] + sPreB[n0 + 32]);
      sZ[mm * D_OUTF + n0 + 48] = (__bf16)(acc3[r] + sPreB[n0 + 48]);
    }
  }
  __syncthreads();

  // ---- GEMM2: Y(64xWC_PAD) = Z x post_w, bias + mask + 1/ola, scatter-add ----
  const int NT2 = WC_PAD / 16;                 // 10 N tiles
  for (int ti = wv; ti < 4 * NT2; ti += 8) {   // 40 tiles, uniform 5 iters/wave
    int m0 = (ti & 3) * 16, n0 = (ti >> 2) * 16;
    v8f acc = {};
    for (int d0 = 0; d0 < D_OUTF; d0 += 32) {
      v16bf a = load_a_frag(sZ + (m0 + lrow) * D_OUTF, d0, lhi);
      v16bf b = *(const v16bf*)(sBpost + (n0 + lrow) * D_OUTF + d0 + lhi * 16);
      acc = __builtin_amdgcn_wmma_f32_16x16x32_bf16(false, a, false, b, (short)0, acc, false, false);
    }
    int o = n0 + lrow;
    if (o < i1) {
      int w = o >> 1, c = o & 1;
      if (sMask[w] != 0.0f) {
        int f = sIdx[w];
        float scale = sOlaInv[w];
        float pb = sPostB[o];
#pragma unroll
        for (int r = 0; r < 8; ++r) {
          int mm = m0 + r + 8 * lhi;
          int rg = row0 + mm, bb = rg >> 9, tt = rg & (T_LEN - 1);
          atomicAdd(out + ((bb * 2 + c) * T_LEN + tt) * F_BINS + f, (acc[r] + pb) * scale);
        }
      }
    }
  }
}

extern "C" void kernel_launch(void* const* d_in, const int* in_sizes, int n_in,
                              void* d_out, int out_size, void* d_ws, size_t ws_size,
                              hipStream_t stream) {
  const float* x      = (const float*)d_in[0];
  const float* pre_w  = (const float*)d_in[1];
  const float* pre_b  = (const float*)d_in[2];
  const float* post_w = (const float*)d_in[3];
  const float* post_b = (const float*)d_in[4];
  const float* mel    = (const float*)d_in[5];
  const float* mask   = (const float*)d_in[6];
  const float* ola    = (const float*)d_in[7];
  const int*   idx    = (const int*)d_in[8];
  float* out = (float*)d_out;

  int i1 = in_sizes[1] / (BANDS * D_OUTF);   // W*C, derived from pre_w size
  if (i1 > WC_PAD) i1 = WC_PAD;
  int W = i1 / 2;

  int zb = (out_size + NTHREADS - 1) / NTHREADS;
  if (zb > 2048) zb = 2048;
  bs_zero_kernel<<<zb, NTHREADS, 0, stream>>>(out, out_size);

  const size_t wpb = (size_t)BANDS * D_OUTF * WC_PAD;      // elems per weight block
  const size_t ws_needed = 2 * wpb * sizeof(__bf16);       // 10.5 MB
  dim3 grid((8 * T_LEN) / TILE_M, BANDS);                  // 64 row tiles x 128 bands

  if (ws_size >= ws_needed && d_ws != nullptr) {
    __bf16* wsBpre  = (__bf16*)d_ws;
    __bf16* wsBpost = wsBpre + wpb;
    bs_prep_weights<<<2048, NTHREADS, 0, stream>>>(pre_w, post_w, wsBpre, wsBpost, i1);
    bandsplit_fused<true><<<grid, NTHREADS, 0, stream>>>(
        x, pre_w, pre_b, post_w, post_b, mel, mask, ola, idx, wsBpre, wsBpost, out, W, i1);
  } else {
    bandsplit_fused<false><<<grid, NTHREADS, 0, stream>>>(
        x, pre_w, pre_b, post_w, post_b, mel, mask, ola, idx, nullptr, nullptr, out, W, i1);
  }
}
